// MultiHeadAttention_27410481283324
// MI455X (gfx1250) — matrix operand order repaired and checked by reading, not compiled, not run on hardware
//
#include <hip/hip_runtime.h>

// Problem constants (match reference)
#define TSEQ  2048
#define DM    768
#define NH    12
#define DHEAD 64
#define BATCH 4
#define BT    (BATCH * TSEQ)   // 8192 rows

typedef __attribute__((ext_vector_type(16))) _Float16 v16h;
typedef __attribute__((ext_vector_type(8)))  _Float16 v8h;
typedef __attribute__((ext_vector_type(8)))  float    v8f;

static __device__ __forceinline__ v8f wmma32(v16h a, v16h b, v8f c) {
  // D = A(16x32 f16) * B(32x16 f16) + C(16x16 f32)
  return __builtin_amdgcn_wmma_f32_16x16x32_f16(false, a, false, b, (short)0, c,
                                                false, false);
}

// A fragment (16x32, f16, row-major source, ld in halves).
// Contiguous k order used here (same as B): lanes 0-15 row M=lane, K = 0..15; lanes 16-31 same
// rows, K = 16..31. VGPR j holds K=2j,2j+1 of its group.
static __device__ __forceinline__ v16h load_a_frag(const _Float16* p, int ld,
                                                   int row0, int k0, int lane) {
  const int r  = lane & 15;
  const int hi = (lane >> 4) & 1;
  const _Float16* base = p + (size_t)(row0 + r) * ld + k0;
  v8h lo = *(const v8h*)(base + hi * 16);
  v8h hv = *(const v8h*)(base + 8 + hi * 16);
  v16h out;
#pragma unroll
  for (int i = 0; i < 8; ++i) { out[i] = lo[i]; out[i + 8] = hv[i]; }
  return out;
}

// B fragment (32x16, f16) where B[k][n] = SRC[n][k] (SRC row-major, ld halves).
// Contiguous k order used here (same as A): lane n holds column n; lanes 0-15 K=0..15,
// lanes 16-31 K=16..31 -> one contiguous 32B load per lane.
static __device__ __forceinline__ v16h load_b_frag(const _Float16* p, int ld,
                                                   int col0, int k0, int lane) {
  const int n  = lane & 15;
  const int hi = (lane >> 4) & 1;
  return *(const v16h*)(p + (size_t)(col0 + n) * ld + k0 + hi * 16);
}

__global__ void __launch_bounds__(256) convert_f32_f16(const float* __restrict__ in,
                                                       _Float16* __restrict__ out,
                                                       int n) {
  int i = blockIdx.x * blockDim.x + threadIdx.x;
  if (i < n) out[i] = (_Float16)in[i];
}

// One wave computes a 32x64 tile of X(8192x768) @ W(768x768)^T (2 A-frags
// sharing 4 B-frags -> 8 WMMAs per k-step), then applies RoPE (for Q,K) in the
// C-fragment layout and scatters f16 results into attention-friendly layouts:
// Q,K -> [B,H,T,DH]; V -> [B,H,DH,T] (transposed).
__global__ void __launch_bounds__(32) qkv_rope_gemm(
    const _Float16* __restrict__ xh,
    const _Float16* __restrict__ wqh, const _Float16* __restrict__ wkh,
    const _Float16* __restrict__ wvh,
    const float* __restrict__ rope_cos, const float* __restrict__ rope_sin,
    _Float16* __restrict__ qbuf, _Float16* __restrict__ kbuf,
    _Float16* __restrict__ vTbuf) {
  const int lane = threadIdx.x;
  const int mt = blockIdx.x;          // 32-row tile of BT
  const int nt = blockIdx.y;          // 64-col tile of DM
  const int w  = blockIdx.z;          // 0=Q 1=K 2=V
  const _Float16* W = (w == 0) ? wqh : (w == 1) ? wkh : wvh;

  v8f acc[2][4] = {};
  for (int k0 = 0; k0 < DM; k0 += 32) {
    if (k0 + 64 < DM) {  // keep L2/WGP$ ahead of the WMMA pipe
      __builtin_prefetch(xh + (size_t)(mt * 32 + (lane & 15)) * DM + k0 + 64, 0, 0);
      __builtin_prefetch(W + (size_t)(nt * 64 + (lane & 15)) * DM + k0 + 64, 0, 0);
    }
    v16h a0 = load_a_frag(xh, DM, mt * 32, k0, lane);
    v16h a1 = load_a_frag(xh, DM, mt * 32 + 16, k0, lane);
#pragma unroll
    for (int j = 0; j < 4; ++j) {
      v16h b = load_b_frag(W, DM, nt * 64 + j * 16, k0, lane);
      acc[0][j] = wmma32(a0, b, acc[0][j]);
      acc[1][j] = wmma32(a1, b, acc[1][j]);
    }
  }

  const int N  = lane & 15;
  const int hi = (lane >> 4) & 1;
#pragma unroll
  for (int r2 = 0; r2 < 2; ++r2) {
#pragma unroll
    for (int j = 0; j < 4; ++j) {
      const int f = nt * 64 + j * 16 + N;   // feature index 0..767
      const int h = f >> 6;
      const int d = f & 63;
#pragma unroll
      for (int i = 0; i < 8; ++i) {
        const int g = mt * 32 + r2 * 16 + i + 8 * hi; // global row in BT
        const int b = g >> 11;                        // / TSEQ
        const int t = g & (TSEQ - 1);
        float v = acc[r2][j][i];
        if (w < 2) {
          // RoPE: feature pairs (2p, 2p+1) sit in adjacent lanes of this frag.
          float other = __shfl_xor(v, 1, 32);
          const int pr = d >> 1;
          const float c = rope_cos[t * 32 + pr];
          const float s = rope_sin[t * 32 + pr];
          v = (d & 1) ? (other * s + v * c) : (v * c - other * s);
          _Float16* dst = (w == 0) ? qbuf : kbuf;
          dst[(((size_t)(b * NH + h)) * TSEQ + t) * DHEAD + d] = (_Float16)v;
        } else {
          // V transposed: [B,H,DH,T] so P*V B-fragments load contiguously.
          vTbuf[(((size_t)(b * NH + h)) * DHEAD + d) * TSEQ + t] = (_Float16)v;
        }
      }
    }
  }
}

// Flash attention: one wave owns 16 queries of one (b,h); iterates 32 keys
// per step with online softmax. 4 WMMAs for S, 4 WMMAs for P*V per step.
__global__ void __launch_bounds__(32) flash_attn(
    const _Float16* __restrict__ qbuf, const _Float16* __restrict__ kbuf,
    const _Float16* __restrict__ vTbuf, _Float16* __restrict__ obuf) {
  const int lane = threadIdx.x;
  const int qt = blockIdx.x;            // 0..T/16-1
  const int bh = blockIdx.y;            // 0..B*H-1
  const int bi = bh / NH, h = bh % NH;
  const int q0 = qt * 16;

  const _Float16* Q = qbuf  + (size_t)bh * TSEQ * DHEAD;
  const _Float16* K = kbuf  + (size_t)bh * TSEQ * DHEAD;
  const _Float16* V = vTbuf + (size_t)bh * DHEAD * TSEQ;

  __shared__ _Float16 plds[16 * 32];    // P tile staging (C-layout -> A-layout)

  // Q A-fragments for dh=0..31 and dh=32..63 (reused across the whole k loop)
  const v16h qa0 = load_a_frag(Q, DHEAD, q0, 0, lane);
  const v16h qa1 = load_a_frag(Q, DHEAD, q0, 32, lane);

  v8f o[4] = {};
  float mrow[8], lrow[8];
#pragma unroll
  for (int i = 0; i < 8; ++i) { mrow[i] = -1e30f; lrow[i] = 0.0f; }

  const int N  = lane & 15;
  const int hi = (lane >> 4) & 1;
  const int kend = q0 + 16;             // exclusive causal key bound

  for (int kb = 0; kb < kend; kb += 32) {
    if (kb + 32 < kend) {               // prefetch next key/value strip
      __builtin_prefetch(K + (size_t)(kb + 32 + N) * DHEAD, 0, 0);
      __builtin_prefetch(V + (size_t)N * TSEQ + kb + 32, 0, 0);
    }
    // ---- S = Q * K^T  (two 16x16 key tiles, K-dim = DHEAD split 2x32) ----
    v8f s0 = {}, s1 = {};
    {
      v16h b00 = *(const v16h*)(K + (size_t)(kb + N) * DHEAD + hi * 16);
      v16h b01 = *(const v16h*)(K + (size_t)(kb + N) * DHEAD + 32 + hi * 16);
      s0 = wmma32(qa0, b00, s0);
      s0 = wmma32(qa1, b01, s0);
      v16h b10 = *(const v16h*)(K + (size_t)(kb + 16 + N) * DHEAD + hi * 16);
      v16h b11 = *(const v16h*)(K + (size_t)(kb + 16 + N) * DHEAD + 32 + hi * 16);
      s1 = wmma32(qa0, b10, s1);
      s1 = wmma32(qa1, b11, s1);
    }

    // ---- online softmax (fp32), per-row stats via 16-lane shuffles ----
#pragma unroll
    for (int i = 0; i < 8; ++i) {
      const int qrow = q0 + i + 8 * hi;
      float a0 = (kb + N      <= qrow) ? s0[i] * 0.125f : -1e30f; // *DH^-0.5
      float a1 = (kb + 16 + N <= qrow) ? s1[i] * 0.125f : -1e30f;
      float mx = fmaxf(a0, a1);
      mx = fmaxf(mx, __shfl_xor(mx, 1, 32));
      mx = fmaxf(mx, __shfl_xor(mx, 2, 32));
      mx = fmaxf(mx, __shfl_xor(mx, 4, 32));
      mx = fmaxf(mx, __shfl_xor(mx, 8, 32));
      const float mnew  = fmaxf(mrow[i], mx);
      const float alpha = __expf(mrow[i] - mnew);
      const float p0 = __expf(a0 - mnew);
      const float p1 = __expf(a1 - mnew);
      float rs = p0 + p1;
      rs += __shfl_xor(rs, 1, 32);
      rs += __shfl_xor(rs, 2, 32);
      rs += __shfl_xor(rs, 4, 32);
      rs += __shfl_xor(rs, 8, 32);
      lrow[i] = lrow[i] * alpha + rs;
      mrow[i] = mnew;
#pragma unroll
      for (int j = 0; j < 4; ++j) o[j][i] *= alpha;
      plds[(i + 8 * hi) * 32 + N]      = (_Float16)p0;
      plds[(i + 8 * hi) * 32 + 16 + N] = (_Float16)p1;
    }
    __syncthreads();

    // ---- re-load P as an A fragment (16x32) ----
    v16h pa;
    {
      const _Float16* base = plds + (lane & 15) * 32;
      v8h lo = *(const v8h*)(base + hi * 16);
      v8h hv = *(const v8h*)(base + 8 + hi * 16);
#pragma unroll
      for (int i = 0; i < 8; ++i) { pa[i] = lo[i]; pa[i + 8] = hv[i]; }
    }
    __syncthreads();

    // ---- O += P * V  (V transposed -> contiguous B fragments) ----
#pragma unroll
    for (int j = 0; j < 4; ++j) {
      v16h vb = *(const v16h*)(V + (size_t)(j * 16 + N) * TSEQ + kb + hi * 16);
      o[j] = wmma32(pa, vb, o[j]);
    }
  }

  // ---- normalize and write O as f16 in [B,T,H*DH] for the final GEMM ----
#pragma unroll
  for (int j = 0; j < 4; ++j) {
#pragma unroll
    for (int i = 0; i < 8; ++i) {
      const float v = o[j][i] / lrow[i];
      const int t = q0 + i + 8 * hi;
      obuf[((size_t)(bi * TSEQ + t)) * DM + h * DHEAD + j * 16 + N] = (_Float16)v;
    }
  }
}

// Final projection: O(8192x768) @ Wo^T -> fp32 output. 32x64 tile per wave.
__global__ void __launch_bounds__(32) out_proj_gemm(
    const _Float16* __restrict__ oh, const _Float16* __restrict__ woh,
    float* __restrict__ out) {
  const int lane = threadIdx.x;
  const int mt = blockIdx.x;
  const int nt = blockIdx.y;

  v8f acc[2][4] = {};
  for (int k0 = 0; k0 < DM; k0 += 32) {
    if (k0 + 64 < DM) {
      __builtin_prefetch(oh + (size_t)(mt * 32 + (lane & 15)) * DM + k0 + 64, 0, 0);
      __builtin_prefetch(woh + (size_t)(nt * 64 + (lane & 15)) * DM + k0 + 64, 0, 0);
    }
    v16h a0 = load_a_frag(oh, DM, mt * 32, k0, lane);
    v16h a1 = load_a_frag(oh, DM, mt * 32 + 16, k0, lane);
#pragma unroll
    for (int j = 0; j < 4; ++j) {
      v16h b = load_b_frag(woh, DM, nt * 64 + j * 16, k0, lane);
      acc[0][j] = wmma32(a0, b, acc[0][j]);
      acc[1][j] = wmma32(a1, b, acc[1][j]);
    }
  }
  const int N  = lane & 15;
  const int hi = (lane >> 4) & 1;
#pragma unroll
  for (int r2 = 0; r2 < 2; ++r2)
#pragma unroll
    for (int j = 0; j < 4; ++j)
#pragma unroll
      for (int i = 0; i < 8; ++i)
        out[(size_t)(mt * 32 + r2 * 16 + i + 8 * hi) * DM + nt * 64 + j * 16 + N] =
            acc[r2][j][i];
}

extern "C" void kernel_launch(void* const* d_in, const int* in_sizes, int n_in,
                              void* d_out, int out_size, void* d_ws, size_t ws_size,
                              hipStream_t stream) {
  const float* x    = (const float*)d_in[0];
  const float* rc   = (const float*)d_in[1];
  const float* rs   = (const float*)d_in[2];
  const float* wq   = (const float*)d_in[3];
  const float* wk   = (const float*)d_in[4];
  const float* wv   = (const float*)d_in[5];
  const float* wo   = (const float*)d_in[6];
  float* out = (float*)d_out;

  // Workspace carve-up (all f16, all 32B-aligned chunk sizes). Total ~68 MB.
  char* p = (char*)d_ws;
  _Float16* xh  = (_Float16*)p; p += (size_t)BT * DM * 2;
  _Float16* wqh = (_Float16*)p; p += (size_t)DM * DM * 2;
  _Float16* wkh = (_Float16*)p; p += (size_t)DM * DM * 2;
  _Float16* wvh = (_Float16*)p; p += (size_t)DM * DM * 2;
  _Float16* woh = (_Float16*)p; p += (size_t)DM * DM * 2;
  _Float16* qb  = (_Float16*)p; p += (size_t)BT * DM * 2;  // [B,H,T,DH]
  _Float16* kbf = (_Float16*)p; p += (size_t)BT * DM * 2;  // [B,H,T,DH]
  _Float16* vT  = (_Float16*)p; p += (size_t)BT * DM * 2;  // [B,H,DH,T]
  _Float16* ob  = (_Float16*)p; p += (size_t)BT * DM * 2;  // [B,T,H*DH]

  const int nX = BT * DM, nW = DM * DM;
  convert_f32_f16<<<(nX + 255) / 256, 256, 0, stream>>>(x,  xh,  nX);
  convert_f32_f16<<<(nW + 255) / 256, 256, 0, stream>>>(wq, wqh, nW);
  convert_f32_f16<<<(nW + 255) / 256, 256, 0, stream>>>(wk, wkh, nW);
  convert_f32_f16<<<(nW + 255) / 256, 256, 0, stream>>>(wv, wvh, nW);
  convert_f32_f16<<<(nW + 255) / 256, 256, 0, stream>>>(wo, woh, nW);

  qkv_rope_gemm<<<dim3(BT / 32, DM / 64, 3), dim3(32), 0, stream>>>(
      xh, wqh, wkh, wvh, rc, rs, qb, kbf, vT);

  flash_attn<<<dim3(TSEQ / 16, BATCH * NH), dim3(32), 0, stream>>>(qb, kbf, vT, ob);

  out_proj_gemm<<<dim3(BT / 32, DM / 64), dim3(32), 0, stream>>>(ob, woh, out);
}
